// GNOKernelLayer_20083267076192
// MI455X (gfx1250) — matrix operand order repaired and checked by reading, not compiled, not run on hardware
//
#include <hip/hip_runtime.h>
#include <math.h>

typedef __attribute__((ext_vector_type(16))) _Float16 v16h;
typedef __attribute__((ext_vector_type(8)))  _Float16 v8h;
typedef __attribute__((ext_vector_type(4)))  _Float16 v4h;
typedef __attribute__((ext_vector_type(8)))  float    v8f;
typedef __attribute__((ext_vector_type(4)))  float    v4f;

// D = A(16x32 f16) * B(32x16 f16) + C(16x16 f32)
static __device__ __forceinline__ v8f wmma16(v16h a, v16h b, v8f c) {
  return __builtin_amdgcn_wmma_f32_16x16x32_f16(false, a, false, b, (short)0, c,
                                                false, false);
}

// ---- A-fragment builders -------------------------------------------------
// 16-bit A 16x32 layout (wave32): lanes 0-15 hold row M=lane, halves[0..7]=K
// kb..kb+7, halves[8..15]=K kb+16..kb+23 with kb = c*32 ; lanes 16-31: +8.
static __device__ __forceinline__ v16h a_frag_row(const float* __restrict__ rowbase,
                                                  int c, int lane) {
  const int lo = c * 32 + ((lane & 16) ? 8 : 0);
  v4f f0 = *(const v4f*)(rowbase + lo);
  v4f f1 = *(const v4f*)(rowbase + lo + 4);
  v4f f2 = *(const v4f*)(rowbase + lo + 16);
  v4f f3 = *(const v4f*)(rowbase + lo + 20);
  v16h a;
#pragma unroll
  for (int i = 0; i < 4; ++i) {
    a[i]      = (_Float16)f0[i];
    a[4 + i]  = (_Float16)f1[i];
    a[8 + i]  = (_Float16)f2[i];
    a[12 + i] = (_Float16)f3[i];
  }
  return a;
}

static __device__ __forceinline__ v16h a_frag_row(const _Float16* __restrict__ rowbase,
                                                  int c, int lane) {
  const int lo = c * 32 + ((lane & 16) ? 8 : 0);
  v8h x = *(const v8h*)(rowbase + lo);
  v8h y = *(const v8h*)(rowbase + lo + 16);
  v16h a;
#pragma unroll
  for (int i = 0; i < 8; ++i) { a[i] = x[i]; a[8 + i] = y[i]; }
  return a;
}

static __device__ __forceinline__ v16h a_frag_from_lds(const _Float16* hrow,
                                                       int c, int lane) {
  const int M  = lane & 15;
  const int lo = c * 32 + ((lane & 16) ? 8 : 0);
  v8h x = *(const v8h*)(hrow + M * 64 + lo);
  v8h y = *(const v8h*)(hrow + M * 64 + lo + 16);
  v16h a;
#pragma unroll
  for (int i = 0; i < 8; ++i) { a[i] = x[i]; a[8 + i] = y[i]; }
  return a;
}

// ---- B-fragment builders -------------------------------------------------
// 16-bit B 32x16 layout (wave32): lanes 0-15: halves[0..7]=K kb..kb+7,
// halves[8..15]=K kb+16..kb+23, N=lane; lanes 16-31: kb offset +8, N=lane-16.
static __device__ __forceinline__ v16h b_frag_plain(const float* __restrict__ W,
                                                    int c, int nb, int lane) {
  const int col = nb * 16 + (lane & 15);
  const int kb  = c * 32 + ((lane & 16) ? 8 : 0);
  v16h b;
#pragma unroll
  for (int t = 0; t < 16; ++t) b[t] = (_Float16)W[(kb + t + ((t & 8) ? 8 : 0)) * 64 + col];
  return b;
}

// W1 is [67][64]; kernel_input ordering is [features(64), rel_pos(3), 0pad]:
// permuted row K<64 -> W1 row K+3 ; K=64..66 -> W1 row K-64 ; else 0.
static __device__ __forceinline__ v16h b_frag_w1(const float* __restrict__ W1,
                                                 int c, int nb, int lane) {
  const int col = nb * 16 + (lane & 15);
  const int kb  = c * 32 + ((lane & 16) ? 8 : 0);
  v16h b;
#pragma unroll
  for (int t = 0; t < 16; ++t) {
    const int K = kb + t + ((t & 8) ? 8 : 0);
    float v;
    if (K < 64)      v = W1[(K + 3) * 64 + col];
    else if (K < 67) v = W1[(K - 64) * 64 + col];
    else             v = 0.0f;
    b[t] = (_Float16)v;
  }
  return b;
}

static __device__ __forceinline__ float gelu_exact(float x) {
  return 0.5f * x * (1.0f + erff(x * 0.70710678118654752f));
}

// ---- features f32 -> f16 pre-convert (bandwidth-trivial) -----------------
__global__ __launch_bounds__(256)
void feat_to_f16(const float* __restrict__ f, _Float16* __restrict__ h, long total4) {
  const long stride = (long)gridDim.x * blockDim.x;
  for (long i = (long)blockIdx.x * blockDim.x + threadIdx.x; i < total4; i += stride) {
    const long e = i * 4;
    v4f x = *(const v4f*)(f + e);
    v4h y;
#pragma unroll
    for (int k = 0; k < 4; ++k) y[k] = (_Float16)x[k];
    *(v4h*)(h + e) = y;
  }
}

// ---- skip kernel: out = features @ Ws + bs  (pure WMMA) ------------------
template <typename FT>
__global__ __launch_bounds__(256, 1)
void gno_skip(const FT* __restrict__ features, const float* __restrict__ Ws,
              const float* __restrict__ bs, float* __restrict__ out, int n) {
  const int lane = threadIdx.x & 31;
  const int wave = threadIdx.x >> 5;
  const int l16  = lane & 15;
  const int half = (lane >> 4) & 1;
  const int tile = blockIdx.x * 8 + wave;       // 16 rows per tile
  if (tile * 16 >= n) return;                   // wave-uniform
  const bool full = (tile * 16 + 16) <= n;

  const int row  = tile * 16 + l16;
  const int rowc = row < n ? row : (n - 1);
  const v16h A0 = a_frag_row(features + (long)rowc * 64, 0, lane);
  const v16h A1 = a_frag_row(features + (long)rowc * 64, 1, lane);

  v16h B[2][4];
#pragma unroll
  for (int c = 0; c < 2; ++c)
#pragma unroll
    for (int nb = 0; nb < 4; ++nb) B[c][nb] = b_frag_plain(Ws, c, nb, lane);

#pragma unroll
  for (int nb = 0; nb < 4; ++nb) {
    const float bv = bs[nb * 16 + l16];
    v8f acc;
#pragma unroll
    for (int r = 0; r < 8; ++r) acc[r] = bv;
    acc = wmma16(A0, B[0][nb], acc);
    acc = wmma16(A1, B[1][nb], acc);
    if (full) {
#pragma unroll
      for (int r = 0; r < 8; ++r) {
        const int orow = tile * 16 + r + half * 8;
        out[(long)orow * 64 + nb * 16 + l16] = acc[r];
      }
    } else {
#pragma unroll
      for (int r = 0; r < 8; ++r) {
        const int orow = tile * 16 + r + half * 8;
        if (orow < n) out[(long)orow * 64 + nb * 16 + l16] = acc[r];
      }
    }
  }
}

// ---- main kernel: one wave == one point ----------------------------------
template <typename FT>
__global__ __launch_bounds__(256, 1)
void gno_main(const float* __restrict__ coords, const FT* __restrict__ features,
              const int* __restrict__ idx, const float* __restrict__ W1,
              const float* __restrict__ b1, const float* __restrict__ W2,
              const float* __restrict__ b2, float* __restrict__ out,
              int n, int waves_total) {
  __shared__ alignas(16) _Float16 hbuf[8][16 * 64];   // per-wave h staging (2KB)
  const int lane = threadIdx.x & 31;
  const int wave = threadIdx.x >> 5;
  const int l16  = lane & 15;
  const int half = (lane >> 4) & 1;
  const int gwave = blockIdx.x * 8 + wave;
  _Float16* hrow = &hbuf[wave][0];

  // Loop-invariant weight fragments, held in VGPRs.
  v16h B1[3][4];
#pragma unroll
  for (int c = 0; c < 3; ++c)
#pragma unroll
    for (int nb = 0; nb < 4; ++nb) B1[c][nb] = b_frag_w1(W1, c, nb, lane);
  v16h B2[2][4];
#pragma unroll
  for (int c = 0; c < 2; ++c)
#pragma unroll
    for (int nb = 0; nb < 4; ++nb) B2[c][nb] = b_frag_plain(W2, c, nb, lane);
  float b1v[4], b2v[4];
#pragma unroll
  for (int nb = 0; nb < 4; ++nb) {
    b1v[nb] = b1[nb * 16 + l16];
    b2v[nb] = b2[nb * 16 + l16];
  }

  const int iters = (n + waves_total - 1) / waves_total;  // uniform trip count
  for (int it = 0; it < iters; ++it) {
    const int p = it * waves_total + gwave;
    const bool valid = p < n;
    // p is wave-uniform: force scalar addressing for center point data
    const int pi = __builtin_amdgcn_readfirstlane(valid ? p : 0);

    // this lane's neighbor row (both lane-halves use the same row)
    const int j = idx[(long)pi * 16 + l16];
    const float cx = coords[(long)pi * 3 + 0];
    const float cy = coords[(long)pi * 3 + 1];
    const float cz = coords[(long)pi * 3 + 2];
    const float rx = coords[(long)j * 3 + 0] - cx;
    const float ry = coords[(long)j * 3 + 1] - cy;
    const float rz = coords[(long)j * 3 + 2] - cz;

    // A fragments straight from global (features) + rel_pos tail chunk
    const FT* frow = features + (long)j * 64;
    const v16h A0 = a_frag_row(frow, 0, lane);
    const v16h A1 = a_frag_row(frow, 1, lane);
    v16h A2;
#pragma unroll
    for (int i = 0; i < 16; ++i) A2[i] = (_Float16)0.0f;
    if (!half) { A2[0] = (_Float16)rx; A2[1] = (_Float16)ry; A2[2] = (_Float16)rz; }

    // GEMM1 + exact GELU -> h (f16) into per-wave LDS tile
#pragma unroll
    for (int nb = 0; nb < 4; ++nb) {
      v8f acc;
#pragma unroll
      for (int r = 0; r < 8; ++r) acc[r] = b1v[nb];
      acc = wmma16(A0, B1[0][nb], acc);
      acc = wmma16(A1, B1[1][nb], acc);
      acc = wmma16(A2, B1[2][nb], acc);
#pragma unroll
      for (int r = 0; r < 8; ++r) {
        const int M = r + half * 8;
        hrow[M * 64 + nb * 16 + l16] = (_Float16)gelu_exact(acc[r]);
      }
    }
    asm volatile("s_wait_dscnt 0" ::: "memory");  // stores -> transposed reload

    const v16h H0 = a_frag_from_lds(hrow, 0, lane);
    const v16h H1 = a_frag_from_lds(hrow, 1, lane);

    // GEMM2 + mean over the 16 neighbor rows
    float sums[4];
#pragma unroll
    for (int nb = 0; nb < 4; ++nb) {
      v8f acc;
#pragma unroll
      for (int r = 0; r < 8; ++r) acc[r] = b2v[nb];
      acc = wmma16(H0, B2[0][nb], acc);
      acc = wmma16(H1, B2[1][nb], acc);
      float s = 0.0f;
#pragma unroll
      for (int r = 0; r < 8; ++r) s += acc[r];     // rows M in this lane-half
      s += __shfl_xor(s, 16, 32);                  // combine both halves
      sums[nb] = s * (1.0f / 16.0f);
    }

    // single guarded RMW block: add mean onto skip result already in out
    if (valid && lane < 16) {
      float* orow = out + (long)pi * 64 + l16;
#pragma unroll
      for (int nb = 0; nb < 4; ++nb) orow[nb * 16] = orow[nb * 16] + sums[nb];
    }
  }
}

extern "C" void kernel_launch(void* const* d_in, const int* in_sizes, int n_in,
                              void* d_out, int out_size, void* d_ws, size_t ws_size,
                              hipStream_t stream) {
  const float* coords   = (const float*)d_in[0];
  const float* features = (const float*)d_in[1];
  const int*   idx      = (const int*)d_in[2];   // int32 per harness convention
  const float* W1 = (const float*)d_in[3];
  const float* b1 = (const float*)d_in[4];
  const float* W2 = (const float*)d_in[5];
  const float* b2 = (const float*)d_in[6];
  const float* Ws = (const float*)d_in[7];
  const float* bs = (const float*)d_in[8];
  float* out = (float*)d_out;

  const int n = in_sizes[0] / 3;   // N from coords [N,3]
  const int tiles   = (n + 15) / 16;
  const int blocksS = (tiles + 7) / 8;
  const int blocksM = 1250;                 // 10000 waves -> uniform iters
  const int wavesT  = blocksM * 8;

  const size_t need_f16 = (size_t)n * 64 * sizeof(_Float16);
  if (ws_size >= need_f16) {
    // pre-convert features to f16 once: halves gather traffic, no in-loop cvt
    _Float16* f16 = (_Float16*)d_ws;
    const long total4 = (long)n * 64 / 4;
    const int blocksC = (int)((total4 + 255) / 256);
    feat_to_f16<<<blocksC, 256, 0, stream>>>(features, f16, total4);
    gno_skip<_Float16><<<blocksS, 256, 0, stream>>>(f16, Ws, bs, out, n);
    gno_main<_Float16><<<blocksM, 256, 0, stream>>>(coords, f16, idx, W1, b1,
                                                    W2, b2, out, n, wavesT);
  } else {
    gno_skip<float><<<blocksS, 256, 0, stream>>>(features, Ws, bs, out, n);
    gno_main<float><<<blocksM, 256, 0, stream>>>(coords, features, idx, W1, b1,
                                                 W2, b2, out, n, wavesT);
  }
}
